// SpatialTransformer_1692217114764
// MI455X (gfx1250) — matrix operand order repaired and checked by reading, not compiled, not run on hardware
//
#include <hip/hip_runtime.h>

// ---------------------------------------------------------------------------
// Types / WMMA helpers (gfx1250, wave32)
// ---------------------------------------------------------------------------
typedef _Float16 v16h __attribute__((ext_vector_type(16)));
typedef _Float16 v8h  __attribute__((ext_vector_type(8)));
typedef float    v8f  __attribute__((ext_vector_type(8)));

__device__ __forceinline__ v8f wmma_f16(v16h a, v16h b, v8f c) {
  // D = A(16x32 f16) * B(32x16 f16) + C(16x16 f32)
  return __builtin_amdgcn_wmma_f32_16x16x32_f16(false, a, false, b, (short)0, c,
                                                false, false);
}

// A-operand fragment: lane holds row M = lane&15, elements K = 16*hi + e
// (one contiguous run of 16 halves). caller passes p = &row[16*hi].
__device__ __forceinline__ v16h load_fragA(const _Float16* p) {
  v8h lo = *(const v8h*)p;
  v8h hi = *(const v8h*)(p + 8);
  return __builtin_shufflevector(lo, hi, 0,1,2,3,4,5,6,7,8,9,10,11,12,13,14,15);
}

// B-operand fragment: lane holds col N = lane&15, elements K = 16*hi + e
// (one contiguous run of 16 halves). caller passes p = &col_run[16*hi].
__device__ __forceinline__ v16h load_fragB(const _Float16* p) {
  v8h lo = *(const v8h*)p;
  v8h hi = *(const v8h*)(p + 8);
  return __builtin_shufflevector(lo, hi, 0,1,2,3,4,5,6,7,8,9,10,11,12,13,14,15);
}

// ---------------------------------------------------------------------------
// CDNA5 async global->LDS copy (ASYNCcnt-tracked; ISA 15.18.3 op 98).
// LDS byte address = low 32 bits of the generic pointer (LDS aperture keeps
// the offset in addr[31:0]).
// ---------------------------------------------------------------------------
__device__ __forceinline__ void async_load_b128(const _Float16* gptr,
                                                const _Float16* lptr) {
  unsigned lds = (unsigned)(uintptr_t)lptr;
  unsigned long long ga = (unsigned long long)(uintptr_t)gptr;
  asm volatile("global_load_async_to_lds_b128 %0, %1, off"
               :: "v"(lds), "v"(ga) : "memory");
}
__device__ __forceinline__ void wait_async0() {
#if __has_builtin(__builtin_amdgcn_s_wait_asynccnt)
  __builtin_amdgcn_s_wait_asynccnt(0);
#else
  asm volatile("s_wait_asynccnt 0x0" ::: "memory");
#endif
}

// ---------------------------------------------------------------------------
// Generic WMMA GEMM: C[M,N] = A[M,K] (f16,row) * Bt[N,K]^T (f16) + bias
// Block tile 128x128, K-step 32, 256 threads = 8 waves (4 along M x 2 along N),
// each wave computes 32x64 (2x4 WMMA tiles). Double-buffered LDS tiles filled
// with async-to-LDS loads that overlap the WMMA work of each iteration.
// out32_mode: 0 none | 1 store f32 | 2 add residual then store f32
// out16_mode: 0 none | 1 store f16 | 2 gelu->f16 | 3 scatter V^T [b,h,d,key]
//             | 4 BCHW f32: outf[b,n,hw] = v + xin[b,n,hw]
// ---------------------------------------------------------------------------
__global__ __launch_bounds__(256) void wmma_gemm(
    const _Float16* __restrict__ A, int lda,
    const _Float16* __restrict__ Bt, int ldb,   // Bt is [N][K], ldb = K
    const float* __restrict__ bias,
    float* __restrict__ out32, int out32_mode,
    _Float16* __restrict__ out16, int out16_mode, int kv_pb,
    float* __restrict__ outf, const float* __restrict__ xin,
    int M, int N, int K)
{
  constexpr int PITCH = 40;  // 32 halves + 8 pad
  __shared__ alignas(16) _Float16 As[2][128 * PITCH];
  __shared__ alignas(16) _Float16 Bs[2][128 * PITCH];  // [n][k]

  const int tid  = threadIdx.x;
  const int wid  = tid >> 5, lane = tid & 31;
  const int lrow = lane & 15, hi = lane >> 4;
  const int waveM = wid & 3, waveN = wid >> 2;
  const int m0 = blockIdx.y * 128, n0 = blockIdx.x * 128;

  // Per-thread tile-fill coordinates: 128 rows x 4 chunks(8 halves) = 512
  // chunks per tile; 256 threads handle rows r and r+64.
  const int rowL = tid >> 2;
  const int kcL  = (tid & 3) * 8;

  v8f acc[2][4];
#pragma unroll
  for (int r = 0; r < 2; ++r)
#pragma unroll
    for (int c = 0; c < 4; ++c) acc[r][c] = (v8f){0,0,0,0,0,0,0,0};

  auto issue_tile = [&](int k0, int buf) {
    async_load_b128(A  + (size_t)(m0 + rowL)      * lda + k0 + kcL,
                    &As[buf][rowL * PITCH + kcL]);
    async_load_b128(A  + (size_t)(m0 + rowL + 64) * lda + k0 + kcL,
                    &As[buf][(rowL + 64) * PITCH + kcL]);
    async_load_b128(Bt + (size_t)(n0 + rowL)      * ldb + k0 + kcL,
                    &Bs[buf][rowL * PITCH + kcL]);
    async_load_b128(Bt + (size_t)(n0 + rowL + 64) * ldb + k0 + kcL,
                    &Bs[buf][(rowL + 64) * PITCH + kcL]);
  };

  issue_tile(0, 0);
  wait_async0();
  __syncthreads();

  int buf = 0;
  for (int k0 = 0; k0 < K; k0 += 32) {
    if (k0 + 32 < K) issue_tile(k0 + 32, buf ^ 1);  // flies during WMMA below

    v16h af[2];
#pragma unroll
    for (int r = 0; r < 2; ++r)
      af[r] = load_fragA(&As[buf][(waveM * 32 + r * 16 + lrow) * PITCH + hi * 16]);
#pragma unroll
    for (int c = 0; c < 4; ++c) {
      v16h bf = load_fragB(&Bs[buf][(waveN * 64 + c * 16 + lrow) * PITCH + hi * 16]);
#pragma unroll
      for (int r = 0; r < 2; ++r) acc[r][c] = wmma_f16(af[r], bf, acc[r][c]);
    }

    wait_async0();
    __syncthreads();
    buf ^= 1;
  }

  // Epilogue. C layout: element e -> row e + 8*hi ; col = lane&15.
  const int mBase = m0 + waveM * 32;
  const int nBase = n0 + waveN * 64;
#pragma unroll
  for (int c = 0; c < 4; ++c) {
    const int n = nBase + c * 16 + lrow;
    const float bv = bias ? bias[n] : 0.0f;
#pragma unroll
    for (int r = 0; r < 2; ++r) {
#pragma unroll
      for (int e = 0; e < 8; ++e) {
        const int m = mBase + r * 16 + e + hi * 8;
        float v = acc[r][c][e] + bv;
        const size_t oi = (size_t)m * N + n;
        if (out32_mode == 2) v += out32[oi];
        if (out32_mode != 0) out32[oi] = v;
        if (out16_mode == 1) {
          out16[oi] = (_Float16)v;
        } else if (out16_mode == 2) {
          out16[oi] = (_Float16)(0.5f * v * (1.0f + erff(v * 0.70710678f)));
        } else if (out16_mode == 3) {
          const int bb = m / kv_pb, key = m - bb * kv_pb;
          const int hh = n >> 6, dd = n & 63;
          out16[(((size_t)bb * 8 + hh) * 64 + dd) * (size_t)kv_pb + key] = (_Float16)v;
        } else if (out16_mode == 4) {
          const int bb = m >> 10, hw = m & 1023;
          const size_t pi = ((size_t)bb * 512 + n) * 1024 + hw;
          outf[pi] = v + xin[pi];
        }
      }
    }
  }
}

// ---------------------------------------------------------------------------
// Flash attention: grid = B*8heads*16 rowblocks, 128 threads = 4 waves,
// each wave owns 16 query rows, online softmax, all matmuls via WMMA.
// Q,Km: [B, rows, 512] f16 (head-major cols). Vt: [B,8,64,KV] f16.
// ---------------------------------------------------------------------------
__global__ __launch_bounds__(128) void flash_attn(
    const _Float16* __restrict__ Q, const _Float16* __restrict__ Km,
    const _Float16* __restrict__ Vt, _Float16* __restrict__ O,
    int KV, int Lvalid, float scale)
{
  __shared__ alignas(16) _Float16 plds[4][16][40];
  const int tid  = threadIdx.x;
  const int wid  = tid >> 5, lane = tid & 31;
  const int lrow = lane & 15, hi = lane >> 4;
  const int blk  = blockIdx.x;
  const int rb = blk & 15, head = (blk >> 4) & 7, b = blk >> 7;
  const int mrow = rb * 64 + wid * 16;

  const _Float16* qp = Q + ((size_t)b * 1024 + mrow + lrow) * 512 + head * 64;
  v16h qf0 = load_fragA(qp + hi * 16);
  v16h qf1 = load_fragA(qp + 32 + hi * 16);

  v8f o0 = (v8f){0,0,0,0,0,0,0,0}, o1 = o0, o2 = o0, o3 = o0;
  float mrun[8], lrun[8];
#pragma unroll
  for (int e = 0; e < 8; ++e) { mrun[e] = -1e30f; lrun[e] = 0.0f; }

  const _Float16* vbase = Vt + ((size_t)(b * 8 + head) * 64) * (size_t)KV;

  for (int j0 = 0; j0 < KV; j0 += 32) {
    // --- S = Q K^T (two 16-col tiles, K-dim 64 = 2 wmma each) ---
    const _Float16* kp = Km + ((size_t)b * KV + j0 + lrow) * 512 + head * 64;
    v16h k00 = load_fragB(kp + hi * 16);
    v16h k01 = load_fragB(kp + 32 + hi * 16);
    v16h k10 = load_fragB(kp + (size_t)16 * 512 + hi * 16);
    v16h k11 = load_fragB(kp + (size_t)16 * 512 + 32 + hi * 16);
    if (j0 + 32 < KV) {   // prefetch next K tile rows
      __builtin_prefetch(kp + (size_t)32 * 512, 0, 1);
      __builtin_prefetch(kp + (size_t)48 * 512, 0, 1);
    }
    v8f s0 = (v8f){0,0,0,0,0,0,0,0}, s1 = s0;
    s0 = wmma_f16(qf0, k00, s0); s0 = wmma_f16(qf1, k01, s0);
    s1 = wmma_f16(qf0, k10, s1); s1 = wmma_f16(qf1, k11, s1);

    const int c0 = j0 + lrow, c1 = j0 + 16 + lrow;
    float f0[8], f1[8], rm[8];
#pragma unroll
    for (int e = 0; e < 8; ++e) {
      f0[e] = (c0 < Lvalid) ? s0[e] * scale : -1e30f;
      f1[e] = (c1 < Lvalid) ? s1[e] * scale : -1e30f;
      rm[e] = fmaxf(f0[e], f1[e]);
    }
    // row max across the 16 lanes sharing each row (half-wave xor reduce)
#pragma unroll
    for (int msk = 1; msk < 16; msk <<= 1)
#pragma unroll
      for (int e = 0; e < 8; ++e) rm[e] = fmaxf(rm[e], __shfl_xor(rm[e], msk, 32));

    float alpha[8], rs[8];
#pragma unroll
    for (int e = 0; e < 8; ++e) {
      const float mn = fmaxf(mrun[e], rm[e]);
      alpha[e] = __expf(mrun[e] - mn);
      mrun[e] = mn;
      f0[e] = __expf(f0[e] - mn);
      f1[e] = __expf(f1[e] - mn);
      rs[e] = f0[e] + f1[e];
    }
#pragma unroll
    for (int msk = 1; msk < 16; msk <<= 1)
#pragma unroll
      for (int e = 0; e < 8; ++e) rs[e] += __shfl_xor(rs[e], msk, 32);
#pragma unroll
    for (int e = 0; e < 8; ++e) {
      lrun[e] = lrun[e] * alpha[e] + rs[e];
      o0[e] *= alpha[e]; o1[e] *= alpha[e]; o2[e] *= alpha[e]; o3[e] *= alpha[e];
    }
    // --- stage P (C-layout -> A-layout) through wave-private LDS tile ---
#pragma unroll
    for (int e = 0; e < 8; ++e) {
      const int row = e + 8 * hi;
      plds[wid][row][lrow]      = (_Float16)f0[e];
      plds[wid][row][16 + lrow] = (_Float16)f1[e];
    }
    v16h pa = load_fragA(&plds[wid][lrow][0] + hi * 16);  // DS in-order per wave
    // --- O += P * V  (4 dim-tiles of 16, K=32 keys) ---
    o0 = wmma_f16(pa, load_fragB(vbase + (size_t)(0*16 + lrow) * KV + j0 + hi*16), o0);
    o1 = wmma_f16(pa, load_fragB(vbase + (size_t)(1*16 + lrow) * KV + j0 + hi*16), o1);
    o2 = wmma_f16(pa, load_fragB(vbase + (size_t)(2*16 + lrow) * KV + j0 + hi*16), o2);
    o3 = wmma_f16(pa, load_fragB(vbase + (size_t)(3*16 + lrow) * KV + j0 + hi*16), o3);
  }

  float inv[8];
#pragma unroll
  for (int e = 0; e < 8; ++e) inv[e] = 1.0f / lrun[e];
  _Float16* op = O + ((size_t)b * 1024 + mrow) * 512 + head * 64;
#pragma unroll
  for (int e = 0; e < 8; ++e) {
    const int row = e + 8 * hi;
    op[(size_t)row * 512 + 0*16 + lrow] = (_Float16)(o0[e] * inv[e]);
    op[(size_t)row * 512 + 1*16 + lrow] = (_Float16)(o1[e] * inv[e]);
    op[(size_t)row * 512 + 2*16 + lrow] = (_Float16)(o2[e] * inv[e]);
    op[(size_t)row * 512 + 3*16 + lrow] = (_Float16)(o3[e] * inv[e]);
  }
}

// ---------------------------------------------------------------------------
// GroupNorm stats: one WG per (b, group): 16 channels x 1024 px contiguous.
// ---------------------------------------------------------------------------
__global__ __launch_bounds__(256) void gn_stats_kernel(const float* __restrict__ x,
                                                       float* __restrict__ stats)
{
  __shared__ float r1[256], r2[256];
  const int bg = blockIdx.x;           // b*32 + g
  const int tid = threadIdx.x;
  const float* p = x + (size_t)bg * 16 * 1024;
  float s = 0.0f, q = 0.0f;
  for (int i = tid; i < 16384; i += 256) { const float v = p[i]; s += v; q += v * v; }
  r1[tid] = s; r2[tid] = q; __syncthreads();
  for (int st = 128; st; st >>= 1) {
    if (tid < st) { r1[tid] += r1[tid + st]; r2[tid] += r2[tid + st]; }
    __syncthreads();
  }
  if (tid == 0) {
    const float mu = r1[0] * (1.0f / 16384.0f);
    const float var = r2[0] * (1.0f / 16384.0f) - mu * mu;
    stats[bg * 2]     = mu;
    stats[bg * 2 + 1] = rsqrtf(var + 1e-5f);
  }
}

// GroupNorm apply + transpose [B,C,HW] -> f16 [B,HW,C] via 32x32 LDS tile.
__global__ __launch_bounds__(1024) void gn_apply_kernel(
    const float* __restrict__ x, const float* __restrict__ stats,
    const float* __restrict__ gam, const float* __restrict__ bet,
    _Float16* __restrict__ out)
{
  __shared__ float t[32][33];
  const int b = blockIdx.z, c0 = blockIdx.y * 32, hw0 = blockIdx.x * 32;
  const int tx = threadIdx.x, ty = threadIdx.y;
  const int c = c0 + ty;
  const float mu = stats[(b * 32 + (c >> 4)) * 2];
  const float rs = stats[(b * 32 + (c >> 4)) * 2 + 1];
  const float v = x[((size_t)b * 512 + c) * 1024 + hw0 + tx];
  t[ty][tx] = (v - mu) * rs * gam[c] + bet[c];
  __syncthreads();
  out[((size_t)b * 1024 + hw0 + ty) * 512 + c0 + tx] = (_Float16)t[tx][ty];
}

// LayerNorm over 512 cols, one row per WG of 128 threads; writes f16.
__global__ __launch_bounds__(128) void ln_kernel(
    const float* __restrict__ h, const float* __restrict__ g,
    const float* __restrict__ bt, _Float16* __restrict__ out)
{
  __shared__ float r1[128], r2[128];
  const int row = blockIdx.x, tid = threadIdx.x;
  const float* p = h + (size_t)row * 512;
  const float a0 = p[tid], a1 = p[tid + 128], a2 = p[tid + 256], a3 = p[tid + 384];
  r1[tid] = a0 + a1 + a2 + a3;
  r2[tid] = a0*a0 + a1*a1 + a2*a2 + a3*a3;
  __syncthreads();
  for (int st = 64; st; st >>= 1) {
    if (tid < st) { r1[tid] += r1[tid + st]; r2[tid] += r2[tid + st]; }
    __syncthreads();
  }
  const float mu = r1[0] * (1.0f / 512.0f);
  const float var = r2[0] * (1.0f / 512.0f) - mu * mu;
  const float rs = rsqrtf(var + 1e-5f);
  _Float16* o = out + (size_t)row * 512;
  o[tid]       = (_Float16)((a0 - mu) * rs * g[tid]       + bt[tid]);
  o[tid + 128] = (_Float16)((a1 - mu) * rs * g[tid + 128] + bt[tid + 128]);
  o[tid + 256] = (_Float16)((a2 - mu) * rs * g[tid + 256] + bt[tid + 256]);
  o[tid + 384] = (_Float16)((a3 - mu) * rs * g[tid + 384] + bt[tid + 384]);
}

// Build zero-padded f16 context [B,96,768] from f32 [B,77,768].
__global__ void ctx_build_kernel(const float* __restrict__ ctx,
                                 _Float16* __restrict__ out, int n)
{
  const int i = blockIdx.x * 256 + threadIdx.x;
  if (i >= n) return;
  const int k = i % 768, r = (i / 768) % 96, b = i / (768 * 96);
  out[i] = (_Float16)(r < 77 ? ctx[((size_t)b * 77 + r) * 768 + k] : 0.0f);
}

// Weight convert f32 [K,N] -> f16 transposed [N,K] via 32x32 LDS tile, so the
// GEMM B-operand loads become contiguous rows (async-to-LDS friendly).
__global__ __launch_bounds__(1024) void wconv_t_kernel(
    const float* __restrict__ src, _Float16* __restrict__ dst, int K, int N)
{
  __shared__ float t[32][33];
  const int k0 = blockIdx.y * 32, n0 = blockIdx.x * 32;
  const int tx = threadIdx.x, ty = threadIdx.y;
  t[ty][tx] = src[(size_t)(k0 + ty) * N + n0 + tx];
  __syncthreads();
  dst[(size_t)(n0 + ty) * K + k0 + tx] = (_Float16)t[tx][ty];
}

// ---------------------------------------------------------------------------
// Host orchestration
// ---------------------------------------------------------------------------
extern "C" void kernel_launch(void* const* d_in, const int* in_sizes, int n_in,
                              void* d_out, int out_size, void* d_ws, size_t ws_size,
                              hipStream_t stream)
{
  (void)in_sizes; (void)n_in; (void)out_size; (void)ws_size;
  const float* x          = (const float*)d_in[0];
  const float* context    = (const float*)d_in[1];
  const float* gn_g       = (const float*)d_in[2];
  const float* gn_b       = (const float*)d_in[3];
  const float* proj_in_w  = (const float*)d_in[4];
  const float* proj_in_b  = (const float*)d_in[5];
  const float* ln1_g      = (const float*)d_in[6];
  const float* ln1_b      = (const float*)d_in[7];
  const float* q1_w       = (const float*)d_in[8];
  const float* k1_w       = (const float*)d_in[9];
  const float* v1_w       = (const float*)d_in[10];
  const float* o1_w       = (const float*)d_in[11];
  const float* o1_b       = (const float*)d_in[12];
  const float* ln2_g      = (const float*)d_in[13];
  const float* ln2_b      = (const float*)d_in[14];
  const float* q2_w       = (const float*)d_in[15];
  const float* k2_w       = (const float*)d_in[16];
  const float* v2_w       = (const float*)d_in[17];
  const float* o2_w       = (const float*)d_in[18];
  const float* o2_b       = (const float*)d_in[19];
  const float* ln3_g      = (const float*)d_in[20];
  const float* ln3_b      = (const float*)d_in[21];
  const float* ff1_w      = (const float*)d_in[22];
  const float* ff1_b      = (const float*)d_in[23];
  const float* ff2_w      = (const float*)d_in[24];
  const float* ff2_b      = (const float*)d_in[25];
  const float* proj_out_w = (const float*)d_in[26];
  const float* proj_out_b = (const float*)d_in[27];

  // ---- workspace carve-up (~122 MB) ----
  char* base = (char*)d_ws; size_t off = 0;
  auto alloc = [&](size_t bytes) -> void* {
    void* p = base + off; off = (off + bytes + 255) & ~(size_t)255; return p;
  };
  float*    h32    = (float*)   alloc(8192ull * 512 * 4);     // residual stream
  _Float16* gn16   = (_Float16*)alloc(8192ull * 512 * 2);
  _Float16* a16    = (_Float16*)alloc(8192ull * 512 * 2);     // LN outputs
  _Float16* q16    = (_Float16*)alloc(8192ull * 512 * 2);
  _Float16* k16s   = (_Float16*)alloc(8192ull * 512 * 2);
  _Float16* vts    = (_Float16*)alloc(8192ull * 512 * 2);     // [b,h,64,1024]
  _Float16* attn16 = (_Float16*)alloc(8192ull * 512 * 2);
  _Float16* ctx16  = (_Float16*)alloc(8ull * 96 * 768 * 2);   // padded context
  _Float16* k16c   = (_Float16*)alloc(8ull * 96 * 512 * 2);
  _Float16* vtc    = (_Float16*)alloc(8ull * 8 * 64 * 96 * 2);
  _Float16* ffa16  = (_Float16*)alloc(8192ull * 2048 * 2);
  _Float16* h16    = (_Float16*)alloc(8192ull * 512 * 2);
  float*    stats  = (float*)   alloc(256 * 2 * 4);
  // weights, all stored transposed [N][K] f16
  _Float16* w_pi  = (_Float16*)alloc(512ull * 512 * 2);
  _Float16* w_q1  = (_Float16*)alloc(512ull * 512 * 2);
  _Float16* w_k1  = (_Float16*)alloc(512ull * 512 * 2);
  _Float16* w_v1  = (_Float16*)alloc(512ull * 512 * 2);
  _Float16* w_o1  = (_Float16*)alloc(512ull * 512 * 2);
  _Float16* w_q2  = (_Float16*)alloc(512ull * 512 * 2);
  _Float16* w_k2  = (_Float16*)alloc(768ull * 512 * 2);
  _Float16* w_v2  = (_Float16*)alloc(768ull * 512 * 2);
  _Float16* w_o2  = (_Float16*)alloc(512ull * 512 * 2);
  _Float16* w_f1  = (_Float16*)alloc(512ull * 2048 * 2);
  _Float16* w_f2  = (_Float16*)alloc(2048ull * 512 * 2);
  _Float16* w_po  = (_Float16*)alloc(512ull * 512 * 2);

  auto convT = [&](const float* s, _Float16* d, int K_, int N_) {
    wconv_t_kernel<<<dim3(N_ / 32, K_ / 32), dim3(32, 32), 0, stream>>>(s, d, K_, N_);
  };
  convT(proj_in_w, w_pi, 512, 512);
  convT(q1_w, w_q1, 512, 512);   convT(k1_w, w_k1, 512, 512);
  convT(v1_w, w_v1, 512, 512);   convT(o1_w, w_o1, 512, 512);
  convT(q2_w, w_q2, 512, 512);   convT(k2_w, w_k2, 768, 512);
  convT(v2_w, w_v2, 768, 512);   convT(o2_w, w_o2, 512, 512);
  convT(ff1_w, w_f1, 512, 2048); convT(ff2_w, w_f2, 2048, 512);
  convT(proj_out_w, w_po, 512, 512);

  auto gemm = [&](const _Float16* A, int lda, const _Float16* B, int ldb,
                  const float* bias, float* o32, int m32, _Float16* o16, int m16,
                  int kvpb, float* ofp, const float* xi, int M, int N, int K) {
    wmma_gemm<<<dim3(N / 128, M / 128), dim3(256), 0, stream>>>(
        A, lda, B, ldb, bias, o32, m32, o16, m16, kvpb, ofp, xi, M, N, K);
  };

  // 1) GroupNorm -> gn16 [B*HW, C]
  gn_stats_kernel<<<dim3(256), dim3(256), 0, stream>>>(x, stats);
  gn_apply_kernel<<<dim3(32, 16, 8), dim3(32, 32), 0, stream>>>(x, stats, gn_g, gn_b, gn16);

  // 2) proj_in -> h32
  gemm(gn16, 512, w_pi, 512, proj_in_b, h32, 1, nullptr, 0, 0, nullptr, nullptr,
       8192, 512, 512);

  // 3) self-attention block
  ln_kernel<<<dim3(8192), dim3(128), 0, stream>>>(h32, ln1_g, ln1_b, a16);
  gemm(a16, 512, w_q1, 512, nullptr, nullptr, 0, q16, 1, 0, nullptr, nullptr, 8192, 512, 512);
  gemm(a16, 512, w_k1, 512, nullptr, nullptr, 0, k16s, 1, 0, nullptr, nullptr, 8192, 512, 512);
  gemm(a16, 512, w_v1, 512, nullptr, nullptr, 0, vts, 3, 1024, nullptr, nullptr, 8192, 512, 512);
  flash_attn<<<dim3(1024), dim3(128), 0, stream>>>(q16, k16s, vts, attn16, 1024, 1024, 0.125f);
  gemm(attn16, 512, w_o1, 512, o1_b, h32, 2, nullptr, 0, 0, nullptr, nullptr, 8192, 512, 512);

  // 4) cross-attention block (keys padded 77 -> 96 with zeros)
  ln_kernel<<<dim3(8192), dim3(128), 0, stream>>>(h32, ln2_g, ln2_b, a16);
  gemm(a16, 512, w_q2, 512, nullptr, nullptr, 0, q16, 1, 0, nullptr, nullptr, 8192, 512, 512);
  ctx_build_kernel<<<dim3((8 * 96 * 768 + 255) / 256), dim3(256), 0, stream>>>(
      context, ctx16, 8 * 96 * 768);
  gemm(ctx16, 768, w_k2, 768, nullptr, nullptr, 0, k16c, 1, 0, nullptr, nullptr, 768, 512, 768);
  gemm(ctx16, 768, w_v2, 768, nullptr, nullptr, 0, vtc, 3, 96, nullptr, nullptr, 768, 512, 768);
  flash_attn<<<dim3(1024), dim3(128), 0, stream>>>(q16, k16c, vtc, attn16, 96, 77, 0.125f);
  gemm(attn16, 512, w_o2, 512, o2_b, h32, 2, nullptr, 0, 0, nullptr, nullptr, 8192, 512, 512);

  // 5) feed-forward (GELU exact)
  ln_kernel<<<dim3(8192), dim3(128), 0, stream>>>(h32, ln3_g, ln3_b, a16);
  gemm(a16, 512, w_f1, 512, ff1_b, nullptr, 0, ffa16, 2, 0, nullptr, nullptr, 8192, 2048, 512);
  gemm(ffa16, 2048, w_f2, 2048, ff2_b, h32, 2, h16, 1, 0, nullptr, nullptr, 8192, 512, 2048);

  // 6) proj_out + x_in -> d_out [B,C,H,W] (f32)
  gemm(h16, 512, w_po, 512, proj_out_b, nullptr, 0, nullptr, 4, 0,
       (float*)d_out, x, 8192, 512, 512);
}
